// OptimizedMultiAttentionHead_1967095022097
// MI455X (gfx1250) — matrix operand order repaired and checked by reading, not compiled, not run on hardware
//
#include <hip/hip_runtime.h>
#include <hip/hip_bf16.h>
#include <math.h>
#include <stdint.h>

// ---------------------------------------------------------------------------
// Problem constants (match reference)
// ---------------------------------------------------------------------------
constexpr int Bc = 2;
constexpr int Tc = 2048;
constexpr int Cc = 1024;
constexpr int Hc = 16;
constexpr int HD = 64;            // head dim
constexpr int Mrows = Bc * Tc;    // 4096

typedef __attribute__((ext_vector_type(16))) __bf16 v16bf;
typedef __attribute__((ext_vector_type(8)))  __bf16 v8bf;
typedef __attribute__((ext_vector_type(8)))  float  v8f;

#define WMMA_BF16(a, b, c) \
  __builtin_amdgcn_wmma_f32_16x16x32_bf16(false, (a), false, (b), (short)0, (c), false, false)

#define NEG_INF (-__builtin_inff())

// ---------------------------------------------------------------------------
// CDNA5 async copy: global memory -> LDS, 16 bytes per lane, ASYNCcnt-tracked
// (GLOBAL_LOAD_ASYNC_TO_LDS_B128, CDNA5 ISA).
// vdst = per-lane LDS byte address (low 32 bits of generic pointer per the
// ISA aperture truncation rule), vaddr = 64-bit global address (GV mode).
// ---------------------------------------------------------------------------
static __device__ inline void async_copy_b128(const void* gptr, void* lptr) {
  unsigned lds = (unsigned)(uintptr_t)lptr;
  unsigned long long ga = (unsigned long long)(uintptr_t)gptr;
  asm volatile("global_load_async_to_lds_b128 %0, %1, off"
               :: "v"(lds), "v"(ga) : "memory");
}
static __device__ inline void wait_async() {
  asm volatile("s_wait_asynccnt 0x0" ::: "memory");
}

// ---------------------------------------------------------------------------
// Fragment loaders (wave32 WMMA operands; contiguous k order used here, same for A and B)
// A 16x32 bf16: lane l -> row = l&15, K = kg*16 + i (contiguous 16 values), kg = l>>4.
// B 32x16 bf16: lane l -> col = l&15, K = kg*16 + i (contiguous 16 values).
// ---------------------------------------------------------------------------
static __device__ inline v16bf load_a_frag(const __bf16* rowbase, int kg) {
  v8bf lo = *(const v8bf*)(rowbase + kg * 16);
  v8bf hi = *(const v8bf*)(rowbase + kg * 16 + 8);
  v16bf a;
#pragma unroll
  for (int i = 0; i < 8; ++i) { a[i] = lo[i]; a[i + 8] = hi[i]; }
  return a;
}

static __device__ inline v16bf load_b_frag(const __bf16* colbase) {
  v8bf lo = *(const v8bf*)(colbase);
  v8bf hi = *(const v8bf*)(colbase + 8);
  v16bf b;
#pragma unroll
  for (int i = 0; i < 8; ++i) { b[i] = lo[i]; b[i + 8] = hi[i]; }
  return b;
}

// ---------------------------------------------------------------------------
// Kernel 1: fp32 -> bf16 conversion; transpose weights to [N][K].
// ---------------------------------------------------------------------------
__global__ __launch_bounds__(256)
void cvt_kernel(const float* __restrict__ x, const float* __restrict__ wqkv,
                const float* __restrict__ wproj,
                __bf16* __restrict__ xb, __bf16* __restrict__ wqkvt,
                __bf16* __restrict__ wprojt) {
  const int NX  = Mrows * Cc;
  const int NW1 = Cc * 3 * Cc;
  const int NW2 = Cc * Cc;
  const int total = NX + NW1 + NW2;
  for (int i = blockIdx.x * blockDim.x + threadIdx.x; i < total;
       i += gridDim.x * blockDim.x) {
    if (i < NX) {
      xb[i] = (__bf16)x[i];
    } else if (i < NX + NW1) {
      int j = i - NX;
      int n = j / Cc, k = j - n * Cc;
      wqkvt[j] = (__bf16)wqkv[(size_t)k * (3 * Cc) + n];
    } else {
      int j = i - NX - NW1;
      int n = j / Cc, k = j - n * Cc;
      wprojt[j] = (__bf16)wproj[(size_t)k * Cc + n];
    }
  }
}

// ---------------------------------------------------------------------------
// Shared tiled-GEMM body: 256 threads = 8 waves; tile 128(M) x 128(N), K-step
// 32; async global->LDS staging; per wave: 1 A-frag + 8 B-frags -> 8 WMMAs.
// acc must be v8f[8].
// ---------------------------------------------------------------------------
#define GEMM_MAIN_LOOP(Aglobal, Bglobal)                                        \
  for (int kk = 0; kk < Cc; kk += 32) {                                         \
    const __bf16* srcA = (Aglobal) + (size_t)(m0 + lr) * Cc + kk + lh * 16;     \
    const __bf16* srcB = (Bglobal) + (size_t)(n0 + lr) * Cc + kk + lh * 16;     \
    __builtin_prefetch(srcA + 32, 0, 1);                                        \
    __builtin_prefetch(srcB + 32, 0, 1);                                        \
    async_copy_b128(srcA,     &As[lr * 40 + lh * 16]);                          \
    async_copy_b128(srcA + 8, &As[lr * 40 + lh * 16 + 8]);                      \
    async_copy_b128(srcB,     &Bs[lr * 40 + lh * 16]);                          \
    async_copy_b128(srcB + 8, &Bs[lr * 40 + lh * 16 + 8]);                      \
    wait_async();                                                               \
    __syncthreads();                                                            \
    v16bf a = load_a_frag(&As[(wave * 16 + ln) * 40], kg);                      \
    v16bf bfr[8];                                                               \
    _Pragma("unroll")                                                           \
    for (int ng = 0; ng < 8; ++ng)                                              \
      bfr[ng] = load_b_frag(&Bs[(ng * 16 + ln) * 40 + kg * 16]);                \
    _Pragma("unroll")                                                           \
    for (int ng = 0; ng < 8; ++ng)                                              \
      acc[ng] = WMMA_BF16(a, bfr[ng], acc[ng]);                                 \
    __syncthreads();                                                            \
  }

// ---------------------------------------------------------------------------
// Kernel 2: QKV GEMM [4096 x 1024] @ [1024 x 3072] (B pre-transposed).
// Scatters into head-major Q (pre-scaled by 1/sqrt(HD)), K, and V^T.
// ---------------------------------------------------------------------------
__global__ __launch_bounds__(256)
void gemm_qkv_kernel(const __bf16* __restrict__ xb, const __bf16* __restrict__ wt,
                     __bf16* __restrict__ qbuf, __bf16* __restrict__ kbuf,
                     __bf16* __restrict__ vtbuf) {
  __shared__ __bf16 As[128 * 40];   // ld 40: rows 16B-aligned + bank padding
  __shared__ __bf16 Bs[128 * 40];
  const int tid  = threadIdx.x;
  const int lane = tid & 31, wave = tid >> 5;
  const int kg = lane >> 4, ln = lane & 15;
  const int m0 = blockIdx.x * 128;
  const int n0 = blockIdx.y * 128;
  const int lr = tid >> 1, lh = tid & 1;   // loaders: 2 threads/row, 32B each

  v8f acc[8] = {};
  GEMM_MAIN_LOOP(xb, wt)

  // Epilogue scatter to head-major q/k/vT (bf16)
#pragma unroll
  for (int ng = 0; ng < 8; ++ng) {
    const int ngl   = n0 + ng * 16 + ln;
    const int which = ngl >> 10;          // 0=q,1=k,2=v
    const int c     = ngl & 1023;
    const int h = c >> 6, d = c & 63;
#pragma unroll
    for (int j = 0; j < 8; ++j) {
      const int mg = m0 + wave * 16 + j + kg * 8;
      const int bb = mg >> 11, t = mg & 2047;
      const int bh = bb * Hc + h;
      const float v = acc[ng][j];
      if (which == 0)
        qbuf[((size_t)bh * Tc + t) * HD + d] = (__bf16)(v * 0.125f); // fold scale
      else if (which == 1)
        kbuf[((size_t)bh * Tc + t) * HD + d] = (__bf16)v;
      else
        vtbuf[((size_t)bh * HD + d) * Tc + t] = (__bf16)v;           // transposed
    }
  }
}

// ---------------------------------------------------------------------------
// Kernel 3: per-KEY-column softmax stats (softmax over the QUERY axis).
// ---------------------------------------------------------------------------
__global__ __launch_bounds__(256)
void attn_colstats_kernel(const __bf16* __restrict__ qb, const __bf16* __restrict__ kb,
                          float* __restrict__ mstat, float* __restrict__ lstat) {
  const int bh  = blockIdx.x;
  const int kc0 = blockIdx.y * 64;
  const int tid = threadIdx.x, lane = tid & 31, wave = tid >> 5;
  const int kg = lane >> 4, ln = lane & 15;
  const __bf16* Qb = qb + (size_t)bh * Tc * HD;
  const __bf16* Kb = kb + (size_t)bh * Tc * HD;

  // K^T B-fragments for this 64-column strip (resident for whole kernel)
  v16bf kfr[4][2];
#pragma unroll
  for (int ng = 0; ng < 4; ++ng)
#pragma unroll
    for (int ks = 0; ks < 2; ++ks)
      kfr[ng][ks] = load_b_frag(&Kb[(size_t)(kc0 + ng * 16 + ln) * HD + ks * 32 + kg * 16]);

  float mcol[4], scol[4];
#pragma unroll
  for (int ng = 0; ng < 4; ++ng) { mcol[ng] = NEG_INF; scol[ng] = 0.f; }

  const int qt0 = kc0 >> 4;                       // causal: only q-tiles >= kc0
  for (int qt = qt0 + wave; qt < Tc / 16; qt += 8) {
    const __bf16* qrow = &Qb[(size_t)(qt * 16 + ln) * HD];
    v16bf a0 = load_a_frag(qrow, kg);
    v16bf a1 = load_a_frag(qrow + 32, kg);
    v8f s4[4];
#pragma unroll
    for (int ng = 0; ng < 4; ++ng) {
      v8f c = {};
      c = WMMA_BF16(a0, kfr[ng][0], c);
      s4[ng] = WMMA_BF16(a1, kfr[ng][1], c);
    }
#pragma unroll
    for (int ng = 0; ng < 4; ++ng) {
      const int col = kc0 + ng * 16 + ln;
      float sv[8], tm = NEG_INF;
#pragma unroll
      for (int j = 0; j < 8; ++j) {
        const int q = qt * 16 + j + kg * 8;
        sv[j] = (q >= col) ? s4[ng][j] : NEG_INF;  // causal mask (keep k<=q)
        tm = fmaxf(tm, sv[j]);
      }
      if (tm > NEG_INF) {                          // online update
        const float mn = fmaxf(mcol[ng], tm);
        const float f  = (mcol[ng] == NEG_INF) ? 0.f : __expf(mcol[ng] - mn);
        float ssum = 0.f;
#pragma unroll
        for (int j = 0; j < 8; ++j)
          if (sv[j] > NEG_INF) ssum += __expf(sv[j] - mn);
        scol[ng] = scol[ng] * f + ssum;
        mcol[ng] = mn;
      }
    }
  }

  __shared__ float sm[8][64];
  __shared__ float ss[8][64];
#pragma unroll
  for (int ng = 0; ng < 4; ++ng) {
    const float mo = __shfl_xor(mcol[ng], 16, 32);
    const float so = __shfl_xor(scol[ng], 16, 32);
    const float m = mcol[ng], s = scol[ng];
    const float mn = fmaxf(m, mo);
    const float sn = ((m  == NEG_INF) ? 0.f : s  * __expf(m  - mn)) +
                     ((mo == NEG_INF) ? 0.f : so * __expf(mo - mn));
    if (lane < 16) { sm[wave][ng * 16 + ln] = mn; ss[wave][ng * 16 + ln] = sn; }
  }
  __syncthreads();
  if (tid < 64) {
    float m = NEG_INF, s = 0.f;
#pragma unroll
    for (int w = 0; w < 8; ++w) {
      const float mo = sm[w][tid], so = ss[w][tid];
      const float mn = fmaxf(m, mo);
      s = ((m  == NEG_INF) ? 0.f : s  * __expf(m  - mn)) +
          ((mo == NEG_INF) ? 0.f : so * __expf(mo - mn));
      m = mn;
    }
    mstat[(size_t)bh * Tc + kc0 + tid] = m;
    lstat[(size_t)bh * Tc + kc0 + tid] = 1.0f / s;
  }
}

// ---------------------------------------------------------------------------
// Kernel 4: O = P @ V with P[q,k] = exp(S-m_k)/l_k for k<=q.
// ---------------------------------------------------------------------------
__global__ __launch_bounds__(256)
void attn_out_kernel(const __bf16* __restrict__ qb, const __bf16* __restrict__ kb,
                     const __bf16* __restrict__ vtb,
                     const float* __restrict__ mstat, const float* __restrict__ lstat,
                     __bf16* __restrict__ obuf) {
  const int bh  = blockIdx.x;
  const int q0b = blockIdx.y * 128;
  const int tid = threadIdx.x, lane = tid & 31, wave = tid >> 5;
  const int kg = lane >> 4, ln = lane & 15;
  const int bIdx = bh / Hc, h = bh % Hc;
  const int q0 = q0b + wave * 16;
  const __bf16* Qb = qb  + (size_t)bh * Tc * HD;
  const __bf16* Kb = kb  + (size_t)bh * Tc * HD;
  const __bf16* Vt = vtb + (size_t)bh * HD * Tc;
  const float*  Ms = mstat + (size_t)bh * Tc;
  const float*  Ls = lstat + (size_t)bh * Tc;

  __shared__ __bf16 Pt[8][16 * 72];   // per-wave 16x64 P tile, ld 72

  const __bf16* qrow = &Qb[(size_t)(q0 + ln) * HD];
  v16bf a0 = load_a_frag(qrow, kg);
  v16bf a1 = load_a_frag(qrow + 32, kg);

  v8f o[4] = {};
  const int kend = q0 + 16;                       // causal bound for this tile
  for (int kc = 0; kc < kend; kc += 64) {
    // --- S tiles: batch fragment loads, then WMMA chain ---
    v16bf kb0[4], kb1[4];
#pragma unroll
    for (int ng = 0; ng < 4; ++ng) {
      const __bf16* krow = &Kb[(size_t)(kc + ng * 16 + ln) * HD];
      kb0[ng] = load_b_frag(krow + kg * 16);
      kb1[ng] = load_b_frag(krow + 32 + kg * 16);
    }
    v8f s4[4];
#pragma unroll
    for (int ng = 0; ng < 4; ++ng) {
      v8f c = {};
      c = WMMA_BF16(a0, kb0[ng], c);
      s4[ng] = WMMA_BF16(a1, kb1[ng], c);
    }
    // --- P = exp(S - m_col) / l_col, masked, to per-wave LDS ---
#pragma unroll
    for (int ng = 0; ng < 4; ++ng) {
      const int col = kc + ng * 16 + ln;
      const float mc = Ms[col];
      const float il = Ls[col];
#pragma unroll
      for (int j = 0; j < 8; ++j) {
        const int q = q0 + j + kg * 8;
        const float p = (q >= col) ? __expf(s4[ng][j] - mc) * il : 0.f;
        Pt[wave][(j + kg * 8) * 72 + ng * 16 + ln] = (__bf16)p;
      }
    }
    asm volatile("s_wait_dscnt 0x0" ::: "memory");  // wave-local transpose fence

    // --- P @ V ---
    v16bf p0 = load_a_frag(&Pt[wave][ln * 72], kg);
    v16bf p1 = load_a_frag(&Pt[wave][ln * 72 + 32], kg);
    v16bf vb0[4], vb1[4];
#pragma unroll
    for (int nd = 0; nd < 4; ++nd) {
      const __bf16* vrow = &Vt[(size_t)(nd * 16 + ln) * Tc + kc];
      vb0[nd] = load_b_frag(vrow + kg * 16);
      vb1[nd] = load_b_frag(vrow + 32 + kg * 16);
    }
#pragma unroll
    for (int nd = 0; nd < 4; ++nd) {
      o[nd] = WMMA_BF16(p0, vb0[nd], o[nd]);
      o[nd] = WMMA_BF16(p1, vb1[nd], o[nd]);
    }
  }

  // Write O row-major [B*T][C] (bf16) for the projection GEMM
#pragma unroll
  for (int nd = 0; nd < 4; ++nd) {
#pragma unroll
    for (int j = 0; j < 8; ++j) {
      const int t = q0 + j + kg * 8;
      obuf[((size_t)(bIdx * Tc + t)) * Cc + h * HD + nd * 16 + ln] = (__bf16)o[nd][j];
    }
  }
}

// ---------------------------------------------------------------------------
// Kernel 5: proj GEMM [4096 x 1024] @ [1024 x 1024] (B transposed) + gate by x.
// ---------------------------------------------------------------------------
__global__ __launch_bounds__(256)
void gemm_proj_kernel(const __bf16* __restrict__ ob, const __bf16* __restrict__ wt,
                      const float* __restrict__ x, float* __restrict__ out) {
  __shared__ __bf16 As[128 * 40];
  __shared__ __bf16 Bs[128 * 40];
  const int tid  = threadIdx.x;
  const int lane = tid & 31, wave = tid >> 5;
  const int kg = lane >> 4, ln = lane & 15;
  const int m0 = blockIdx.x * 128;
  const int n0 = blockIdx.y * 128;
  const int lr = tid >> 1, lh = tid & 1;

  v8f acc[8] = {};
  GEMM_MAIN_LOOP(ob, wt)

#pragma unroll
  for (int ng = 0; ng < 8; ++ng) {
#pragma unroll
    for (int j = 0; j < 8; ++j) {
      const int mg = m0 + wave * 16 + j + kg * 8;
      const int nc = n0 + ng * 16 + ln;
      const size_t idx = (size_t)mg * Cc + nc;
      out[idx] = x[idx] * acc[ng][j];              // multiplicative gate
    }
  }
}

// ---------------------------------------------------------------------------
// Host launch
// ---------------------------------------------------------------------------
extern "C" void kernel_launch(void* const* d_in, const int* in_sizes, int n_in,
                              void* d_out, int out_size, void* d_ws, size_t ws_size,
                              hipStream_t stream) {
  (void)in_sizes; (void)n_in; (void)out_size; (void)ws_size;
  const float* x     = (const float*)d_in[0];
  const float* wqkv  = (const float*)d_in[1];
  const float* wproj = (const float*)d_in[2];
  float* out = (float*)d_out;

  char* ws = (char*)d_ws;
  size_t off = 0;
  auto alloc = [&](size_t bytes) -> void* {
    void* p = ws + off;
    off += (bytes + 255) & ~(size_t)255;
    return p;
  };
  __bf16* xb     = (__bf16*)alloc((size_t)Mrows * Cc * 2);        // 8 MB
  __bf16* wqkvt  = (__bf16*)alloc((size_t)3 * Cc * Cc * 2);       // 6 MB
  __bf16* wprojt = (__bf16*)alloc((size_t)Cc * Cc * 2);           // 2 MB
  __bf16* qbuf   = (__bf16*)alloc((size_t)Bc * Hc * Tc * HD * 2); // 8 MB
  __bf16* kbuf   = (__bf16*)alloc((size_t)Bc * Hc * Tc * HD * 2); // 8 MB
  __bf16* vtbuf  = (__bf16*)alloc((size_t)Bc * Hc * HD * Tc * 2); // 8 MB
  __bf16* obuf   = (__bf16*)alloc((size_t)Mrows * Cc * 2);        // 8 MB
  float*  mstat  = (float*)alloc((size_t)Bc * Hc * Tc * 4);       // 256 KB
  float*  lstat  = (float*)alloc((size_t)Bc * Hc * Tc * 4);       // 256 KB

  cvt_kernel<<<4096, 256, 0, stream>>>(x, wqkv, wproj, xb, wqkvt, wprojt);
  gemm_qkv_kernel<<<dim3(Mrows / 128, (3 * Cc) / 128), 256, 0, stream>>>(
      xb, wqkvt, qbuf, kbuf, vtbuf);
  attn_colstats_kernel<<<dim3(Bc * Hc, Tc / 64), 256, 0, stream>>>(
      qbuf, kbuf, mstat, lstat);
  attn_out_kernel<<<dim3(Bc * Hc, Tc / 128), 256, 0, stream>>>(
      qbuf, kbuf, vtbuf, mstat, lstat, obuf);
  gemm_proj_kernel<<<dim3(Mrows / 128, Cc / 128), 256, 0, stream>>>(
      obuf, wprojt, x, out);
}
